// TransformerEncoder_13194139533726
// MI455X (gfx1250) — matrix operand order repaired and checked by reading, not compiled, not run on hardware
//
#include <hip/hip_runtime.h>
#include <hip/hip_bf16.h>

// ---------------------------------------------------------------------------
// Problem constants (from reference): B=64 samples, N=64 patches, INF=ATT=768,
// HID=3072, ROWS = B*N = 4096 flattened token rows.
// ---------------------------------------------------------------------------
static const int C_ROWS = 4096;   // B * N
static const int C_INF  = 768;
static const int C_ATT  = 768;
static const int C_HID  = 3072;

typedef __attribute__((ext_vector_type(16))) __bf16 v16bf;
typedef __attribute__((ext_vector_type(8)))  __bf16 v8bf;
typedef __attribute__((ext_vector_type(8)))  float  v8f;

#define LDS_STRIDE 40   // bf16 elements per LDS tile row (80B: 16B aligned, conflict-free)

union V16U { v16bf v; v8bf h[2]; };

__device__ inline v16bf load_frag(const __bf16* base, int secondHalfOff) {
  V16U u;
  u.h[0] = *(const v8bf*)(base);
  u.h[1] = *(const v8bf*)(base + secondHalfOff);
  return u.v;
}

__device__ inline v8f wmma_bf16(v16bf a, v16bf b, v8f c) {
  return __builtin_amdgcn_wmma_f32_16x16x32_bf16(false, a, false, b, (short)0, c, false, false);
}

__device__ inline v8f vzero8() { v8f z = {}; return z; }

// Async direct global->LDS staging: 32B per thread per matrix (2 x b128),
// ISA applies INST_OFFSET to both the LDS and global addresses.
__device__ inline void async_stage32(unsigned ldsAddr, const __bf16* gptr) {
  asm volatile(
      "global_load_async_to_lds_b128 %0, %1, off\n\t"
      "global_load_async_to_lds_b128 %0, %1, off offset:16"
      :: "v"(ldsAddr), "v"((unsigned long long)gptr)
      : "memory");
}

__device__ inline void wait_async0() {
  asm volatile("s_wait_asynccnt 0x0" ::: "memory");
}

__device__ inline unsigned lds_addr_of(const void* p) {
  // Generic pointers into LDS carry the LDS offset in their low 32 bits
  // (CDNA5 ISA 10.2).
  return (unsigned)(unsigned long long)p;
}

// ---------------------------------------------------------------------------
// Patchify: x[64,3,128,128] -> flat[b, n=i*8+j, f=c*256+pi*16+pj] as bf16.
// ---------------------------------------------------------------------------
__global__ __launch_bounds__(256)
void patchify_kernel(const float* __restrict__ x, __bf16* __restrict__ flatBF) {
  long long idx = (long long)blockIdx.x * blockDim.x + threadIdx.x;
  if (idx >= (long long)C_ROWS * C_INF) return;
  int f   = (int)(idx % C_INF);
  int row = (int)(idx / C_INF);          // b*64 + patch
  int b   = row >> 6, np = row & 63;
  int i   = np >> 3,  j  = np & 7;
  int c   = f / 256,  rem = f % 256;
  int pi  = rem >> 4, pj  = rem & 15;
  float v = x[(((long long)(b * 3 + c)) * 128 + (i * 16 + pi)) * 128 + (j * 16 + pj)];
  flatBF[idx] = (__bf16)v;
}

// ---------------------------------------------------------------------------
// One-time weight prep: Wt[n][k] (bf16) = W[k][n] (f32). 32x32 tiles via LDS so
// both the f32 reads and bf16 writes are coalesced. Grid: (N/32, K/32).
// ---------------------------------------------------------------------------
__global__ __launch_bounds__(256)
void wtrans_kernel(const float* __restrict__ W, __bf16* __restrict__ Wt, int K, int N) {
  __shared__ float tile[32][33];
  const int t  = threadIdx.x;
  const int tx = t & 31, ty = t >> 5;                 // 32 x 8
  const int n0 = blockIdx.x * 32, k0 = blockIdx.y * 32;
#pragma unroll
  for (int i = 0; i < 4; ++i)
    tile[ty + 8 * i][tx] = W[(long long)(k0 + ty + 8 * i) * N + n0 + tx];
  __syncthreads();
#pragma unroll
  for (int i = 0; i < 4; ++i)
    Wt[(long long)(n0 + ty + 8 * i) * K + k0 + tx] = (__bf16)tile[tx][ty + 8 * i];
}

// ---------------------------------------------------------------------------
// Big-tile WMMA GEMM for the MLP trunk: C = A(bf16, MxK) * Bt^T + bias[+addend]
// A row-major [M][K] bf16; Bt pre-transposed [N][K] bf16. Block tile 128x128,
// 8 waves; wave w owns a 16x128 strip: 8 WMMAs per 32-K step sharing one
// A-fragment. Double-buffered LDS; tile i+1 is staged with async
// global->LDS b128 transfers (ASYNCcnt) while tile i is consumed by WMMA.
// All B fragments are fetched before the WMMA chain so the compiler can use
// partial DScnt waits instead of a full drain per WMMA.
// Grid: (N/128, M/128).
// ---------------------------------------------------------------------------
__global__ __launch_bounds__(256)
void gemm128_bf16_kernel(const __bf16* __restrict__ A, int lda,
                         const __bf16* __restrict__ Bt, int ldb,
                         const float* __restrict__ bias,
                         const float* __restrict__ addend, int ldadd,
                         float*  __restrict__ Cf, int ldcf,
                         __bf16* __restrict__ Cb, int ldcb,
                         int K) {
  __shared__ __align__(16) __bf16 As[2][128 * LDS_STRIDE];
  __shared__ __align__(16) __bf16 Bs[2][128 * LDS_STRIDE];

  const int t    = threadIdx.x;
  const int lane = t & 31, w = t >> 5;
  const int h    = lane >> 4, l15 = lane & 15;
  const int n0   = blockIdx.x * 128, m0 = blockIdx.y * 128;
  const int sr   = t >> 1, sk = (t & 1) * 16;   // staging: row, 16-elem k chunk

  // per-thread staging base addresses (advance by k in elements)
  const __bf16* ap = &A[(long long)(m0 + sr) * lda + sk];
  const __bf16* bp = &Bt[(long long)(n0 + sr) * ldb + sk];
  const unsigned laBase[2] = { lds_addr_of(&As[0][sr * LDS_STRIDE + sk]),
                               lds_addr_of(&As[1][sr * LDS_STRIDE + sk]) };
  const unsigned lbBase[2] = { lds_addr_of(&Bs[0][sr * LDS_STRIDE + sk]),
                               lds_addr_of(&Bs[1][sr * LDS_STRIDE + sk]) };

  v8f acc[8];
#pragma unroll
  for (int i = 0; i < 8; ++i) acc[i] = vzero8();

  // prologue: stage tile 0 into buffer 0
  async_stage32(laBase[0], ap);
  async_stage32(lbBase[0], bp);
  wait_async0();
  __syncthreads();

  const int nsteps = K / 32;
  for (int i = 0; i < nsteps; ++i) {
    const int buf = i & 1;
    if (i + 1 < nsteps) {           // kick off next tile into the other buffer
      const int kn = (i + 1) * 32;
      async_stage32(laBase[buf ^ 1], ap + kn);
      async_stage32(lbBase[buf ^ 1], bp + kn);
    }
    const __bf16* asb = As[buf];
    const __bf16* bsb = Bs[buf];
    // fetch all fragments first (DS returns in order -> partial dscnt waits)
    v16bf af = load_frag(&asb[(w * 16 + l15) * LDS_STRIDE + h * 16], 8);
    v16bf bf[8];
#pragma unroll
    for (int sn = 0; sn < 8; ++sn)
      bf[sn] = load_frag(&bsb[(sn * 16 + l15) * LDS_STRIDE + h * 16], 8);
#pragma unroll
    for (int sn = 0; sn < 8; ++sn)
      acc[sn] = wmma_bf16(af, bf[sn], acc[sn]);
    if (i + 1 < nsteps) {
      wait_async0();                // own transfers done; barrier covers the rest
      __syncthreads();
    }
  }

#pragma unroll
  for (int sn = 0; sn < 8; ++sn) {
#pragma unroll
    for (int r = 0; r < 8; ++r) {
      long long row = m0 + w * 16 + h * 8 + r;
      int col = n0 + sn * 16 + l15;
      float v = acc[sn][r] + bias[col];
      if (addend) v += addend[row * (long long)ldadd + col];
      if (Cf) Cf[row * (long long)ldcf + col] = v;
      if (Cb) Cb[row * (long long)ldcb + col] = (__bf16)v;
    }
  }
}

// ---------------------------------------------------------------------------
// 64x64 WMMA GEMM with fused f32->bf16 weight staging. Used for the batched
// per-patch QKV projections (M=64; each weight element read exactly once, so
// streaming-convert beats a pre-pass). Grid: (N/64, M/64, batch).
// ---------------------------------------------------------------------------
__global__ __launch_bounds__(256)
void gemm64_bf16_kernel(const __bf16* __restrict__ A, long long Abatch, int lda,
                        const float*  __restrict__ W, long long Wbatch, int ldw,
                        const float*  __restrict__ bias, long long biasBatch,
                        __bf16* __restrict__ Cb, int ldcb, long long CbBatch,
                        int K) {
  __shared__ __align__(16) __bf16 As[64 * LDS_STRIDE];
  __shared__ __align__(16) __bf16 Bs[64 * LDS_STRIDE];

  const int t    = threadIdx.x;
  const int lane = t & 31, w = t >> 5;
  const int h    = lane >> 4, l15 = lane & 15;
  const int n0   = blockIdx.x * 64, m0 = blockIdx.y * 64;
  const int batch = blockIdx.z;

  A += (long long)batch * Abatch;
  W += (long long)batch * Wbatch;
  const float* biasp = bias + (long long)batch * biasBatch;

  const int sm  = (2 * w) >> 2;
  const int sn0 = (2 * w) & 3, sn1 = sn0 + 1;
  const int ar = t >> 2, ac = (t & 3) * 8;
  const int bk = t >> 3, bc = (t & 7) * 8;

  v8f acc0 = vzero8(), acc1 = vzero8();

  for (int k0 = 0; k0 < K; k0 += 32) {
    __syncthreads();
    *(v8bf*)&As[ar * LDS_STRIDE + ac] =
        *(const v8bf*)&A[(long long)(m0 + ar) * lda + k0 + ac];
    {
      const float* wp = &W[(long long)(k0 + bk) * ldw + n0 + bc];
#pragma unroll
      for (int i = 0; i < 8; ++i)
        Bs[(bc + i) * LDS_STRIDE + bk] = (__bf16)wp[i];
    }
    __syncthreads();
    v16bf af = load_frag(&As[(sm  * 16 + l15) * LDS_STRIDE + h * 16], 8);
    v16bf b0 = load_frag(&Bs[(sn0 * 16 + l15) * LDS_STRIDE + h * 16], 8);
    v16bf b1 = load_frag(&Bs[(sn1 * 16 + l15) * LDS_STRIDE + h * 16], 8);
    acc0 = wmma_bf16(af, b0, acc0);
    acc1 = wmma_bf16(af, b1, acc1);
  }

#pragma unroll
  for (int r = 0; r < 8; ++r) {
    int ml = sm * 16 + h * 8 + r;
    long long row = m0 + ml;
    {
      int col = n0 + sn0 * 16 + l15;
      float v = acc0[r] + biasp[col];
      Cb[(long long)batch * CbBatch + row * (long long)ldcb + col] = (__bf16)v;
    }
    {
      int col = n0 + sn1 * 16 + l15;
      float v = acc1[r] + biasp[col];
      Cb[(long long)batch * CbBatch + row * (long long)ldcb + col] = (__bf16)v;
    }
  }
}

// ---------------------------------------------------------------------------
// Per-patch attention across the batch dim: scores = q k^T / sqrt(768),
// softmax over c, attn = P v.  One block (8 waves) per patch.
// ---------------------------------------------------------------------------
__global__ __launch_bounds__(256)
void attention_kernel(const __bf16* __restrict__ qbf, const __bf16* __restrict__ kbf,
                      const __bf16* __restrict__ vbf, float* __restrict__ attnF) {
  __shared__ __align__(16) __bf16 As[64 * LDS_STRIDE];
  __shared__ __align__(16) __bf16 Bs[64 * LDS_STRIDE];
  __shared__ float  Sc[64 * 65];
  __shared__ __align__(16) __bf16 Pb[64 * 72];

  const int n    = blockIdx.x;
  const int t    = threadIdx.x;
  const int lane = t & 31, w = t >> 5;
  const int h    = lane >> 4, l15 = lane & 15;
  const int sm   = (2 * w) >> 2, sn0 = (2 * w) & 3, sn1 = sn0 + 1;
  const int ar = t >> 2, ac = (t & 3) * 8;
  const int bk = t >> 3, bc = (t & 7) * 8;

  // ---- phase 1: scores (64x64, K = 768) ----
  v8f acc0 = vzero8(), acc1 = vzero8();
  for (int k0 = 0; k0 < 768; k0 += 32) {
    __syncthreads();
    *(v8bf*)&As[ar * LDS_STRIDE + ac] =
        *(const v8bf*)&qbf[((long long)(ar * 64 + n)) * 768 + k0 + ac];
    *(v8bf*)&Bs[ar * LDS_STRIDE + ac] =
        *(const v8bf*)&kbf[((long long)(ar * 64 + n)) * 768 + k0 + ac];
    __syncthreads();
    v16bf af = load_frag(&As[(sm  * 16 + l15) * LDS_STRIDE + h * 16], 8);
    v16bf b0 = load_frag(&Bs[(sn0 * 16 + l15) * LDS_STRIDE + h * 16], 8);
    v16bf b1 = load_frag(&Bs[(sn1 * 16 + l15) * LDS_STRIDE + h * 16], 8);
    acc0 = wmma_bf16(af, b0, acc0);
    acc1 = wmma_bf16(af, b1, acc1);
  }
  const float scale = 0.03608439182435161f;  // 1/sqrt(768)
#pragma unroll
  for (int r = 0; r < 8; ++r) {
    int brow = sm * 16 + h * 8 + r;
    Sc[brow * 65 + sn0 * 16 + l15] = acc0[r] * scale;
    Sc[brow * 65 + sn1 * 16 + l15] = acc1[r] * scale;
  }
  __syncthreads();

  // ---- softmax over c (rows b = 0..63), write P as bf16 ----
  if (t < 64) {
    float mx = -1e30f;
    for (int c = 0; c < 64; ++c) mx = fmaxf(mx, Sc[t * 65 + c]);
    float s = 0.f;
    for (int c = 0; c < 64; ++c) { float e = __expf(Sc[t * 65 + c] - mx); Sc[t * 65 + c] = e; s += e; }
    float inv = 1.f / s;
    for (int c = 0; c < 64; ++c) Pb[t * 72 + c] = (__bf16)(Sc[t * 65 + c] * inv);
  }

  // ---- phase 2: attn = P (64x64) x v (64x768), in 12 column blocks ----
  for (int nb = 0; nb < 12; ++nb) {
    acc0 = vzero8(); acc1 = vzero8();
    for (int kk0 = 0; kk0 < 64; kk0 += 32) {
      __syncthreads();   // also makes Pb visible on first entry
      const __bf16* vp = &vbf[((long long)((kk0 + bk) * 64 + n)) * 768 + nb * 64 + bc];
#pragma unroll
      for (int i = 0; i < 8; ++i)
        Bs[(bc + i) * LDS_STRIDE + bk] = vp[i];
      __syncthreads();
      v16bf af = load_frag(&Pb[(sm * 16 + l15) * 72 + kk0 + h * 16], 8);
      v16bf b0 = load_frag(&Bs[(sn0 * 16 + l15) * LDS_STRIDE + h * 16], 8);
      v16bf b1 = load_frag(&Bs[(sn1 * 16 + l15) * LDS_STRIDE + h * 16], 8);
      acc0 = wmma_bf16(af, b0, acc0);
      acc1 = wmma_bf16(af, b1, acc1);
    }
#pragma unroll
    for (int r = 0; r < 8; ++r) {
      int brow = sm * 16 + h * 8 + r;
      long long row = (long long)brow * 64 + n;
      attnF[row * 768 + nb * 64 + sn0 * 16 + l15] = acc0[r];
      attnF[row * 768 + nb * 64 + sn1 * 16 + l15] = acc1[r];
    }
  }
}

// ---------------------------------------------------------------------------
// Row LayerNorm over 768, in place. One block per row.
// ---------------------------------------------------------------------------
__global__ __launch_bounds__(256)
void layernorm768_kernel(float* __restrict__ a, const float* __restrict__ g,
                         const float* __restrict__ b) {
  __shared__ float red[256];
  const int row = blockIdx.x, t = threadIdx.x;
  float* p = a + (long long)row * 768;
  float x0 = p[t], x1 = p[t + 256], x2 = p[t + 512];
  red[t] = x0 + x1 + x2;
  __syncthreads();
  for (int s = 128; s > 0; s >>= 1) { if (t < s) red[t] += red[t + s]; __syncthreads(); }
  float mu = red[0] * (1.f / 768.f);
  __syncthreads();
  float d0 = x0 - mu, d1 = x1 - mu, d2 = x2 - mu;
  red[t] = d0 * d0 + d1 * d1 + d2 * d2;
  __syncthreads();
  for (int s = 128; s > 0; s >>= 1) { if (t < s) red[t] += red[t + s]; __syncthreads(); }
  float rinv = rsqrtf(red[0] * (1.f / 768.f) + 1e-5f);
  p[t]       = d0 * rinv * g[t]       + b[t];
  p[t + 256] = d1 * rinv * g[t + 256] + b[t + 256];
  p[t + 512] = d2 * rinv * g[t + 512] + b[t + 512];
}

// ---------------------------------------------------------------------------
// Row softmax over 768, in place. One block per row.
// ---------------------------------------------------------------------------
__global__ __launch_bounds__(256)
void softmax768_kernel(float* __restrict__ a) {
  __shared__ float red[256];
  const int row = blockIdx.x, t = threadIdx.x;
  float* p = a + (long long)row * 768;
  float x0 = p[t], x1 = p[t + 256], x2 = p[t + 512];
  red[t] = fmaxf(x0, fmaxf(x1, x2));
  __syncthreads();
  for (int s = 128; s > 0; s >>= 1) { if (t < s) red[t] = fmaxf(red[t], red[t + s]); __syncthreads(); }
  float m = red[0];
  __syncthreads();
  float e0 = __expf(x0 - m), e1 = __expf(x1 - m), e2 = __expf(x2 - m);
  red[t] = e0 + e1 + e2;
  __syncthreads();
  for (int s = 128; s > 0; s >>= 1) { if (t < s) red[t] += red[t + s]; __syncthreads(); }
  float inv = 1.f / red[0];
  p[t] = e0 * inv; p[t + 256] = e1 * inv; p[t + 512] = e2 * inv;
}

// ---------------------------------------------------------------------------
// Host-side launch
// ---------------------------------------------------------------------------
extern "C" void kernel_launch(void* const* d_in, const int* in_sizes, int n_in,
                              void* d_out, int out_size, void* d_ws, size_t ws_size,
                              hipStream_t stream) {
  (void)in_sizes; (void)n_in; (void)out_size; (void)ws_size;
  const float* x    = (const float*)d_in[0];
  const float* Wq   = (const float*)d_in[1];
  const float* bq   = (const float*)d_in[2];
  const float* Wk   = (const float*)d_in[3];
  const float* bkv  = (const float*)d_in[4];
  const float* Wv   = (const float*)d_in[5];
  const float* bv   = (const float*)d_in[6];
  const float* ln_g = (const float*)d_in[7];
  const float* ln_b = (const float*)d_in[8];
  const float* p1w  = (const float*)d_in[9];
  const float* p1b  = (const float*)d_in[10];
  const float* w0   = (const float*)d_in[11];
  const float* b0   = (const float*)d_in[12];
  const float* w1   = (const float*)d_in[13];
  const float* b1   = (const float*)d_in[14];
  const float* w2   = (const float*)d_in[15];
  const float* b2   = (const float*)d_in[16];
  const float* w3   = (const float*)d_in[17];
  const float* b3   = (const float*)d_in[18];
  const float* w4   = (const float*)d_in[19];
  const float* b4   = (const float*)d_in[20];
  const float* p2w  = (const float*)d_in[21];
  const float* p2b  = (const float*)d_in[22];
  float* out = (float*)d_out;

  // Workspace layout (~176 MB)
  char* ws = (char*)d_ws;
  size_t off = 0;
  auto take = [&](size_t bytes) {
    char* p = ws + off;
    off = (off + bytes + 255) & ~(size_t)255;
    return p;
  };
  __bf16* flatBF = (__bf16*)take((size_t)C_ROWS * C_INF * 2);
  __bf16* qbf    = (__bf16*)take((size_t)C_ROWS * C_ATT * 2);
  __bf16* kbf    = (__bf16*)take((size_t)C_ROWS * C_ATT * 2);
  __bf16* vbf    = (__bf16*)take((size_t)C_ROWS * C_ATT * 2);
  float*  attnF  = (float* )take((size_t)C_ROWS * C_ATT * 4);
  __bf16* attBF  = (__bf16*)take((size_t)C_ROWS * C_ATT * 2);
  __bf16* hA     = (__bf16*)take((size_t)C_ROWS * C_HID * 2);
  __bf16* hB     = (__bf16*)take((size_t)C_ROWS * C_HID * 2);
  float*  t1f    = (float* )take((size_t)C_ROWS * C_ATT * 4);
  // pre-transposed bf16 weights [N][K]
  __bf16* w0t = (__bf16*)take((size_t)C_HID * C_ATT * 2);
  __bf16* w1t = (__bf16*)take((size_t)C_HID * C_HID * 2);
  __bf16* w2t = (__bf16*)take((size_t)C_HID * C_HID * 2);
  __bf16* w3t = (__bf16*)take((size_t)C_HID * C_HID * 2);
  __bf16* w4t = (__bf16*)take((size_t)C_ATT * C_HID * 2);
  __bf16* p1t = (__bf16*)take((size_t)C_ATT * C_ATT * 2);
  __bf16* p2t = (__bf16*)take((size_t)C_ATT * C_ATT * 2);

  const long long WQ_STRIDE = (long long)C_INF * C_ATT;  // per-patch weight
  const int LDA_QKV = 64 * C_INF;                        // row stride across b

  // 0) one-time weight transpose+convert for the shared GEMMs
  wtrans_kernel<<<dim3(C_HID / 32, C_ATT / 32), 256, 0, stream>>>(w0, w0t, C_ATT, C_HID);
  wtrans_kernel<<<dim3(C_HID / 32, C_HID / 32), 256, 0, stream>>>(w1, w1t, C_HID, C_HID);
  wtrans_kernel<<<dim3(C_HID / 32, C_HID / 32), 256, 0, stream>>>(w2, w2t, C_HID, C_HID);
  wtrans_kernel<<<dim3(C_HID / 32, C_HID / 32), 256, 0, stream>>>(w3, w3t, C_HID, C_HID);
  wtrans_kernel<<<dim3(C_ATT / 32, C_HID / 32), 256, 0, stream>>>(w4, w4t, C_HID, C_ATT);
  wtrans_kernel<<<dim3(C_ATT / 32, C_INF / 32), 256, 0, stream>>>(p1w, p1t, C_INF, C_ATT);
  wtrans_kernel<<<dim3(C_ATT / 32, C_ATT / 32), 256, 0, stream>>>(p2w, p2t, C_ATT, C_ATT);

  // 1) patchify
  patchify_kernel<<<(C_ROWS * C_INF) / 256, 256, 0, stream>>>(x, flatBF);

  // 2) per-patch q/k/v projections (batched over 64 patches; f32 weights,
  //    streamed once each -> fused convert in the GEMM)
  dim3 gqkv(C_ATT / 64, 1, 64);
  gemm64_bf16_kernel<<<gqkv, 256, 0, stream>>>(flatBF, C_INF, LDA_QKV, Wq, WQ_STRIDE, C_ATT,
                                               bq, C_ATT, qbf, LDA_QKV, C_ATT, C_INF);
  gemm64_bf16_kernel<<<gqkv, 256, 0, stream>>>(flatBF, C_INF, LDA_QKV, Wk, WQ_STRIDE, C_ATT,
                                               bkv, C_ATT, kbf, LDA_QKV, C_ATT, C_INF);
  gemm64_bf16_kernel<<<gqkv, 256, 0, stream>>>(flatBF, C_INF, LDA_QKV, Wv, WQ_STRIDE, C_ATT,
                                               bv, C_ATT, vbf, LDA_QKV, C_ATT, C_INF);

  // 3) attention across batch dim, per patch
  attention_kernel<<<64, 256, 0, stream>>>(qbf, kbf, vbf, attnF);

  // 4) LayerNorm in place
  layernorm768_kernel<<<C_ROWS, 256, 0, stream>>>(attnF, ln_g, ln_b);

  // 5) att = ln + flat @ p1_w + p1_b   (bf16 out only)
  gemm128_bf16_kernel<<<dim3(C_ATT / 128, C_ROWS / 128), 256, 0, stream>>>(
      flatBF, C_INF, p1t, C_INF, p1b, attnF, C_ATT,
      nullptr, 0, attBF, C_ATT, C_INF);

  // 6) linear stack
  gemm128_bf16_kernel<<<dim3(C_HID / 128, C_ROWS / 128), 256, 0, stream>>>(
      attBF, C_ATT, w0t, C_ATT, b0, nullptr, 0,
      nullptr, 0, hA, C_HID, C_ATT);
  gemm128_bf16_kernel<<<dim3(C_HID / 128, C_ROWS / 128), 256, 0, stream>>>(
      hA, C_HID, w1t, C_HID, b1, nullptr, 0,
      nullptr, 0, hB, C_HID, C_HID);
  gemm128_bf16_kernel<<<dim3(C_HID / 128, C_ROWS / 128), 256, 0, stream>>>(
      hB, C_HID, w2t, C_HID, b2, nullptr, 0,
      nullptr, 0, hA, C_HID, C_HID);
  gemm128_bf16_kernel<<<dim3(C_HID / 128, C_ROWS / 128), 256, 0, stream>>>(
      hA, C_HID, w3t, C_HID, b3, nullptr, 0,
      nullptr, 0, hB, C_HID, C_HID);

  // 7) t1 = att @ p2_w + p2_b  (f32)
  gemm128_bf16_kernel<<<dim3(C_ATT / 128, C_ROWS / 128), 256, 0, stream>>>(
      attBF, C_ATT, p2t, C_ATT, p2b, nullptr, 0,
      t1f, C_ATT, nullptr, 0, C_ATT);

  // 8) proj = h3 @ w4 + b4 + t1  -> d_out (f32)
  gemm128_bf16_kernel<<<dim3(C_ATT / 128, C_ROWS / 128), 256, 0, stream>>>(
      hB, C_HID, w4t, C_HID, b4, t1f, C_ATT,
      out, C_ATT, nullptr, 0, C_HID);

  // 9) final softmax over last dim, in place on d_out
  softmax768_kernel<<<C_ROWS, 256, 0, stream>>>(out);
}
